// IMUFullWindowEncoder_63617055588822
// MI455X (gfx1250) — matrix operand order repaired and checked by reading, not compiled, not run on hardware
//
#include <hip/hip_runtime.h>
#include <math.h>

// ---------------- CDNA5 WMMA plumbing (wave32, 16x16x32 bf16, fp32 acc) ----
typedef __bf16 bf16_t;
typedef __attribute__((ext_vector_type(16))) __bf16 v16bf;
typedef __attribute__((ext_vector_type(8)))  __bf16 v8bf;
typedef __attribute__((ext_vector_type(8)))  float  v8f;

#define BATCH   32
#define T_LEN   512
#define NSUB    503      // T - SUBWIN + 1
#define SEQ_LEN 504      // cls + NSUB
#define EMB     768
#define LH      128
#define G4      512      // 4 * LSTM_H

__device__ __forceinline__ float sigmoidf_(float x){ return 1.f/(1.f+__expf(-x)); }

__device__ __forceinline__ v8f zero8(){
  v8f z;
#pragma unroll
  for (int i = 0; i < 8; ++i) z[i] = 0.f;
  return z;
}

// A fragment: 16x32 bf16 tile, row-major in LDS with row stride ld (elements).
// ISA layout is contiguous in 16B chunks per lane group:
//   lanes 0-15 : halves 0..7 = K0..7   (bytes 0..15),  halves 8..15 = K16..23 (bytes 32..47)
//   lanes 16-31: halves 0..7 = K8..15  (bytes 16..31), halves 8..15 = K24..31 (bytes 48..63)
// => two 128-bit LDS loads per lane, no shuffles. Requires 16B-aligned base, ld % 8 == 0.
__device__ __forceinline__ v16bf load_a16(const bf16_t* a, int ld){
  const int lane = threadIdx.x & 31;
  const int m = lane & 15, g = lane >> 4;
  const bf16_t* row = a + m*ld + g*8;
  union { v16bf v; v8bf h[2]; } u;
  u.h[0] = *(const v8bf*)(row);        // K 8g .. 8g+7
  u.h[1] = *(const v8bf*)(row + 16);   // K 16+8g .. 16+8g+7
  return u.v;
}

// B fragment from tile-fragment-packed global memory: each 32Kx16N tile is 512
// bf16 (1024B) stored per-lane: element (lane, half) at tile[lane*16 + half],
// lane<16 -> (n=lane, K 0..7,16..23), lane>=16 -> (n=lane-16, K 8..15,24..31).
// => two coalesced 128-bit global loads per lane.
__device__ __forceinline__ v16bf load_b16p(const bf16_t* tile){
  const int lane = threadIdx.x & 31;
  const bf16_t* p = tile + lane*16;
  union { v16bf v; v8bf h[2]; } u;
  u.h[0] = *(const v8bf*)(p);
  u.h[1] = *(const v8bf*)(p + 8);
  return u.v;
}

__device__ __forceinline__ v8f wmma_bf16(v16bf a, v16bf b, v8f c){
  return __builtin_amdgcn_wmma_f32_16x16x32_bf16(false, a, false, b, (short)0, c, false, false);
}

// ---------------- prep kernels ---------------------------------------------
// Pack W[n][k] (row-major fp32, N x K) into fragment-packed bf16 tiles,
// tile order (n_tile * (K/32) + k_chunk), 512 elements per tile.
__device__ __forceinline__ void pack_index(int i, int K, int& n, int& k){
  int KT = K >> 5;
  int tile = i >> 9, rem = i & 511;
  int lane = rem >> 4, ii = rem & 15;
  int nt = tile / KT, kc = tile % KT;
  n = nt*16 + (lane & 15);
  k = kc*32 + ((ii >> 3) << 4) + ((lane >> 4) << 3) + (ii & 7);
}

__global__ void pack_b_bf16_kernel(const float* __restrict__ src, bf16_t* __restrict__ dst,
                                   int N, int K){
  int i = blockIdx.x*256 + threadIdx.x;
  if (i >= N*K) return;
  int n, k; pack_index(i, K, n, k);
  dst[i] = (bf16_t)src[(size_t)n*K + k];
}

// Combined LSTM weight [Wih | Whh] -> packed tiles (K=256, N=512), bias sum
__global__ void prep_lstm_w_kernel(const float* __restrict__ wih, const float* __restrict__ whh,
                                   const float* __restrict__ bih, const float* __restrict__ bhh,
                                   bf16_t* __restrict__ wcat, float* __restrict__ bias){
  int i = blockIdx.x*256 + threadIdx.x;
  if (i < 256*512){
    int n, k; pack_index(i, 256, n, k);
    float v = (k < LH) ? wih[(size_t)n*LH + k] : whh[(size_t)n*LH + (k - LH)];
    wcat[i] = (bf16_t)v;
  }
  if (i < 512) bias[i] = bih[i] + bhh[i];
}

__global__ void cls_fill_kernel(const float* __restrict__ cls, float* __restrict__ seq){
  int i = blockIdx.x*256 + threadIdx.x;
  if (i >= BATCH*EMB) return;
  int b = i / EMB, e = i % EMB;
  seq[(size_t)b*SEQ_LEN*EMB + e] = cls[e];
}

// ---------------- subwindow encoder ----------------------------------------
struct EncParams {
  const float *conv1_w, *conv1_b, *bn1_g, *bn1_b, *bn1_m, *bn1_v;
  const float *conv2_w, *conv2_b, *bn2_g, *bn2_b, *bn2_m, *bn2_v;
  const float *attn_w, *attn_b, *fc_b;
  const bf16_t *wcat_f; const float *bias_f;
  const bf16_t *wcat_b; const float *bias_b;
  const bf16_t *fc_t;                 // packed tiles, K=256, N=768
};

// One workgroup = 16 subwindows (WMMA M dim). 8 wave32 waves.
// LDS (byte offsets, overlapped by live range):
//   [0      ) s_x    bf16[16][10][128]   conv stack output (A operand)
//   [40960  ) s_h1   f32 [16][10][64]    conv1 out   | s_out bf16[16][10][256]
//   [122880 ) s_raw  f32 [16][12][3]     | s_gates f32[16][512] | s_pool bf16[16][256]
//   [155648 ) s_h    bf16[16][128], [159744) s_c f32[16][128], [167936) s_red f32[16][10]
__global__ __launch_bounds__(256)
void encoder_kernel(const float* __restrict__ inp, EncParams p,
                    const float* __restrict__ acc_weight, int is_acc,
                    float* __restrict__ seq){
  __shared__ char smem[168576] __attribute__((aligned(16)));
  bf16_t* s_x     = (bf16_t*)(smem);
  float*  s_h1    = (float* )(smem + 40960);
  bf16_t* s_out   = (bf16_t*)(smem + 40960);
  float*  s_raw   = (float* )(smem + 122880);
  float*  s_gates = (float* )(smem + 122880);
  bf16_t* s_pool  = (bf16_t*)(smem + 122880);
  bf16_t* s_h     = (bf16_t*)(smem + 155648);
  float*  s_c     = (float* )(smem + 159744);
  float*  s_red   = (float* )(smem + 167936);

  const int tid  = threadIdx.x;
  const int lane = tid & 31;
  const int wave = tid >> 5;
  const int sw0  = blockIdx.x * 16;

  // ---- load raw windows (time padded by 1 each side for conv1) ----
  for (int i = tid; i < 16*12*3; i += 256){
    int ch = i % 3, t = (i/3) % 12, sw = i / 36;
    int swg = sw0 + sw, b = swg / NSUB, n = swg % NSUB;
    float v = 0.f;
    if (t >= 1 && t <= 10) v = inp[((size_t)b*T_LEN + (n + t - 1))*3 + ch];
    s_raw[sw*36 + t*3 + ch] = v;
  }
  __syncthreads();

  // ---- conv1 (3->64, k3 pad1) + relu + bn1 ----
  for (int i = tid; i < 16*10*64; i += 256){
    int c = i & 63, t = (i >> 6) % 10, sw = i / 640;
    float a = p.conv1_b[c];
#pragma unroll
    for (int k = 0; k < 3; ++k)
#pragma unroll
      for (int ci = 0; ci < 3; ++ci)
        a += p.conv1_w[(c*3 + ci)*3 + k] * s_raw[sw*36 + (t + k)*3 + ci];
    a = fmaxf(a, 0.f);
    float sc = p.bn1_g[c] * rsqrtf(p.bn1_v[c] + 1e-5f);
    s_h1[sw*640 + t*64 + c] = (a - p.bn1_m[c]) * sc + p.bn1_b[c];
  }
  __syncthreads();

  // ---- conv2 (64->128, k3 pad1) + relu + bn2 -> bf16 ----
  for (int i = tid; i < 16*10*128; i += 256){
    int c = i & 127, t = (i >> 7) % 10, sw = i / 1280;
    float a = p.conv2_b[c];
    const float* w  = p.conv2_w + (size_t)c*192;
    const float* h1 = s_h1 + sw*640 + t*64;
    for (int ci = 0; ci < 64; ++ci){
      float x0 = (t > 0) ? h1[ci - 64] : 0.f;
      float x2 = (t < 9) ? h1[ci + 64] : 0.f;
      a += w[ci*3+0]*x0 + w[ci*3+1]*h1[ci] + w[ci*3+2]*x2;
    }
    a = fmaxf(a, 0.f);
    float sc = p.bn2_g[c] * rsqrtf(p.bn2_v[c] + 1e-5f);
    s_x[sw*1280 + t*128 + c] = (bf16_t)((a - p.bn2_m[c]) * sc + p.bn2_b[c]);
  }
  __syncthreads();

  // ---- BiLSTM: gates = [x_t | h] (16x256) @ Wcat (256x512) via WMMA ----
  for (int dir = 0; dir < 2; ++dir){
    const bf16_t* W   = dir ? p.wcat_b : p.wcat_f;
    const float* bias = dir ? p.bias_b : p.bias_f;
    for (int i = tid; i < 16*128; i += 256){ s_h[i] = (bf16_t)0.f; s_c[i] = 0.f; }
    __syncthreads();
    for (int step = 0; step < 10; ++step){
      int t = dir ? (9 - step) : step;
      v8f acc[4];
#pragma unroll
      for (int j = 0; j < 4; ++j) acc[j] = zero8();
#pragma unroll
      for (int kc = 0; kc < 8; ++kc){
        v16bf a = (kc < 4) ? load_a16(s_x + t*128 + kc*32, 1280)
                           : load_a16(s_h + (kc - 4)*32, 128);
#pragma unroll
        for (int j = 0; j < 4; ++j){
          v16bf b = load_b16p(W + (size_t)((wave*4 + j)*8 + kc)*512);
          acc[j] = wmma_bf16(a, b, acc[j]);
        }
      }
      {
        int n = lane & 15, mb = (lane >> 4) * 8;
#pragma unroll
        for (int j = 0; j < 4; ++j){
          int col = (wave*4 + j)*16 + n;
          float bc = bias[col];
#pragma unroll
          for (int r = 0; r < 8; ++r) s_gates[(mb + r)*G4 + col] = acc[j][r] + bc;
        }
      }
      __syncthreads();
      // elementwise LSTM cell update (i,f,g,o order per jnp.split)
      for (int i = tid; i < 16*128; i += 256){
        int sw = i >> 7, j = i & 127;
        float ig = sigmoidf_(s_gates[sw*G4 + j]);
        float fg = sigmoidf_(s_gates[sw*G4 + 128 + j]);
        float gg = tanhf   (s_gates[sw*G4 + 256 + j]);
        float og = sigmoidf_(s_gates[sw*G4 + 384 + j]);
        float c  = fg * s_c[i] + ig * gg;
        float h  = og * tanhf(c);
        s_c[i] = c; s_h[i] = (bf16_t)h;
        s_out[sw*2560 + t*256 + dir*128 + j] = (bf16_t)h;
      }
      __syncthreads();
    }
  }

  // ---- attention pooling over 10 timesteps ----
  for (int i = tid; i < 160; i += 256){
    int sw = i / 10, t = i % 10;
    float s = p.attn_b[0];
    const bf16_t* o = s_out + sw*2560 + t*256;
    for (int c = 0; c < 256; ++c) s += (float)o[c] * p.attn_w[c];
    s_red[sw*10 + t] = s;
  }
  __syncthreads();
  if (tid < 16){
    float mx = -1e30f;
    for (int t = 0; t < 10; ++t) mx = fmaxf(mx, s_red[tid*10 + t]);
    float se = 0.f;
    for (int t = 0; t < 10; ++t){ float e = __expf(s_red[tid*10+t] - mx); s_red[tid*10+t] = e; se += e; }
    float inv = 1.f/se;
    for (int t = 0; t < 10; ++t) s_red[tid*10+t] *= inv;
  }
  __syncthreads();
  for (int i = tid; i < 16*256; i += 256){
    int sw = i >> 8, c = i & 255;
    float a = 0.f;
    for (int t = 0; t < 10; ++t) a += s_red[sw*10+t] * (float)s_out[sw*2560 + t*256 + c];
    s_pool[i] = (bf16_t)a;
  }
  __syncthreads();

  // ---- fc: pooled (16x256) @ fc_w^T (256x768) via WMMA; fused merge-scale ----
  float aw = acc_weight[0];
  float sig = sigmoidf_(aw);
  float scale = is_acc ? sig : (1.f - sig);
  v8f accf[6];
#pragma unroll
  for (int j = 0; j < 6; ++j) accf[j] = zero8();
#pragma unroll
  for (int kc = 0; kc < 8; ++kc){
    v16bf a = load_a16(s_pool + kc*32, 256);
#pragma unroll
    for (int j = 0; j < 6; ++j){
      v16bf b = load_b16p(p.fc_t + (size_t)((wave*6 + j)*8 + kc)*512);
      accf[j] = wmma_bf16(a, b, accf[j]);
    }
  }
  {
    int n = lane & 15, mb = (lane >> 4) * 8;
#pragma unroll
    for (int j = 0; j < 6; ++j){
      int col = (wave*6 + j)*16 + n;
      float bc = p.fc_b[col];
#pragma unroll
      for (int r = 0; r < 8; ++r){
        int swg = sw0 + mb + r;
        int bb = swg / NSUB, nn = swg % NSUB;
        float val = (accf[j][r] + bc) * scale;
        size_t off = ((size_t)bb*SEQ_LEN + nn + 1)*EMB + col;
        if (is_acc) seq[off] = val;      // acc pass writes
        else        seq[off] += val;     // gyro pass accumulates (stream-ordered)
      }
    }
  }
}

// ---------------- K/V projection GEMM: (B*504 x 768) @ (768 x 1536) --------
__global__ __launch_bounds__(256)
void kv_gemm_kernel(const float* __restrict__ seq, const bf16_t* __restrict__ kvw,
                    const float* __restrict__ in_proj_b, bf16_t* __restrict__ kv){
  __shared__ bf16_t sA[16*768] __attribute__((aligned(16)));
  const int tid = threadIdx.x, lane = tid & 31, wave = tid >> 5;
  const int m0 = blockIdx.x * 16;
  const int nb = blockIdx.y * 512;
  for (int i = tid; i < 16*768; i += 256)
    sA[i] = (bf16_t)seq[(size_t)(m0 + i/768)*EMB + (i % 768)];
  __syncthreads();
  v8f acc[4];
#pragma unroll
  for (int j = 0; j < 4; ++j) acc[j] = zero8();
  for (int kc = 0; kc < 24; ++kc){
    v16bf a = load_a16(sA + kc*32, 768);
#pragma unroll
    for (int j = 0; j < 4; ++j){
      int nt = blockIdx.y*32 + wave*4 + j;       // global n-tile (of 96)
      v16bf b = load_b16p(kvw + (size_t)(nt*24 + kc)*512);
      acc[j] = wmma_bf16(a, b, acc[j]);
    }
  }
  const float* kvb = in_proj_b + EMB;   // k/v bias slice
  int n = lane & 15, mb = (lane >> 4) * 8;
#pragma unroll
  for (int j = 0; j < 4; ++j){
    int col = nb + wave*64 + j*16 + n;
    float bc = kvb[col];
#pragma unroll
    for (int r = 0; r < 8; ++r)
      kv[(size_t)(m0 + mb + r)*1536 + col] = (bf16_t)(acc[j][r] + bc);
  }
}

// ---------------- CLS-row attention per (head, batch) ----------------------
__global__ __launch_bounds__(256)
void attn_cls_kernel(const float* __restrict__ seq, const bf16_t* __restrict__ kv,
                     const float* __restrict__ in_proj_w, const float* __restrict__ in_proj_b,
                     float* __restrict__ attn_cls){
  __shared__ float q[96];
  __shared__ float sc[SEQ_LEN];
  const int h = blockIdx.x, b = blockIdx.y, tid = threadIdx.x;
  const float* cls = seq + (size_t)b*SEQ_LEN*EMB;
  for (int d = tid; d < 96; d += 256){
    int g = h*96 + d;
    float a = in_proj_b[g];
    const float* w = in_proj_w + (size_t)g*EMB;
    for (int e = 0; e < EMB; ++e) a += w[e]*cls[e];
    q[d] = a;
  }
  __syncthreads();
  for (int s = tid; s < SEQ_LEN; s += 256){
    const bf16_t* kr = kv + (size_t)(b*SEQ_LEN + s)*1536 + h*96;
    float a = 0.f;
    for (int d = 0; d < 96; ++d) a += q[d]*(float)kr[d];
    sc[s] = a * 0.10206207f;   // 1/sqrt(96)
  }
  __syncthreads();
  if (tid == 0){
    float mx = -1e30f;
    for (int s = 0; s < SEQ_LEN; ++s) mx = fmaxf(mx, sc[s]);
    float se = 0.f;
    for (int s = 0; s < SEQ_LEN; ++s){ float e = __expf(sc[s]-mx); sc[s]=e; se+=e; }
    float inv = 1.f/se;
    for (int s = 0; s < SEQ_LEN; ++s) sc[s] *= inv;
  }
  __syncthreads();
  for (int d = tid; d < 96; d += 256){
    float a = 0.f;
    for (int s = 0; s < SEQ_LEN; ++s)
      a += sc[s] * (float)kv[(size_t)(b*SEQ_LEN + s)*1536 + EMB + h*96 + d];
    attn_cls[(size_t)b*EMB + h*96 + d] = a;
  }
}

// ---------------- small (32 x 768) @ (768 x 768)^T projections -------------
__global__ void proj_kernel(const float* __restrict__ x, const float* __restrict__ W,
                            const float* __restrict__ bias, float* __restrict__ y){
  int i = blockIdx.x*256 + threadIdx.x;
  if (i >= BATCH*EMB) return;
  int b = i / EMB, e = i % EMB;
  float a = bias[e];
  const float* w  = W + (size_t)e*EMB;
  const float* xr = x + (size_t)b*EMB;
  for (int k = 0; k < EMB; ++k) a += w[k]*xr[k];
  y[i] = a;
}

// ---------------- host launch ----------------------------------------------
extern "C" void kernel_launch(void* const* d_in, const int* in_sizes, int n_in,
                              void* d_out, int out_size, void* d_ws, size_t ws_size,
                              hipStream_t stream){
  (void)in_sizes; (void)n_in; (void)out_size; (void)ws_size;
  // setup_inputs() dict order: acc, gyro, acc_params(24), gyro_params(24), head_params(8)
  const int ACC_P = 2, GYR_P = 26, HEAD = 50;
  const float* acc  = (const float*)d_in[0];
  const float* gyro = (const float*)d_in[1];

  char* wp = (char*)d_ws;
  auto carve = [&](size_t bytes) -> char* {
    char* r = wp; wp += (bytes + 255) & ~(size_t)255; return r;
  };
  float*  seq  = (float*) carve((size_t)BATCH*SEQ_LEN*EMB*4);      // merged seq (fp32)
  bf16_t* kv   = (bf16_t*)carve((size_t)BATCH*SEQ_LEN*1536*2);     // K|V rows (bf16)
  bf16_t* wcat[4]; float* biasv[4];
  for (int i = 0; i < 4; ++i){
    wcat[i]  = (bf16_t*)carve((size_t)256*512*2);
    biasv[i] = (float*) carve((size_t)512*4);
  }
  bf16_t* fct[2];
  for (int i = 0; i < 2; ++i) fct[i] = (bf16_t*)carve((size_t)256*768*2);
  bf16_t* kvw      = (bf16_t*)carve((size_t)768*1536*2);
  float*  attn_cls = (float*) carve((size_t)BATCH*EMB*4);
  float*  tmp      = (float*) carve((size_t)BATCH*EMB*4);

  // weight prep (fragment-packed bf16 tiles)
  for (int s = 0; s < 2; ++s){
    int base = s ? GYR_P : ACC_P;
    for (int d = 0; d < 2; ++d){
      prep_lstm_w_kernel<<<512, 256, 0, stream>>>(
          (const float*)d_in[base + 12 + d*4 + 0], (const float*)d_in[base + 12 + d*4 + 1],
          (const float*)d_in[base + 12 + d*4 + 2], (const float*)d_in[base + 12 + d*4 + 3],
          wcat[s*2 + d], biasv[s*2 + d]);
    }
    pack_b_bf16_kernel<<<768, 256, 0, stream>>>(
        (const float*)d_in[base + 22], fct[s], 768, 256);          // fc_w (768x256)
  }
  pack_b_bf16_kernel<<<4608, 256, 0, stream>>>(
      ((const float*)d_in[HEAD + 2]) + (size_t)768*768, kvw, 1536, 768); // k|v rows of in_proj_w
  cls_fill_kernel<<<96, 256, 0, stream>>>((const float*)d_in[HEAD + 1], seq);

  // encoders: acc writes w*e, gyro accumulates (1-w)*e  (B*N = 16096 = 1006*16)
  for (int s = 0; s < 2; ++s){
    int base = s ? GYR_P : ACC_P;
    EncParams p;
    p.conv1_w = (const float*)d_in[base+0];  p.conv1_b = (const float*)d_in[base+1];
    p.bn1_g   = (const float*)d_in[base+2];  p.bn1_b   = (const float*)d_in[base+3];
    p.bn1_m   = (const float*)d_in[base+4];  p.bn1_v   = (const float*)d_in[base+5];
    p.conv2_w = (const float*)d_in[base+6];  p.conv2_b = (const float*)d_in[base+7];
    p.bn2_g   = (const float*)d_in[base+8];  p.bn2_b   = (const float*)d_in[base+9];
    p.bn2_m   = (const float*)d_in[base+10]; p.bn2_v   = (const float*)d_in[base+11];
    p.attn_w  = (const float*)d_in[base+20]; p.attn_b  = (const float*)d_in[base+21];
    p.fc_b    = (const float*)d_in[base+23];
    p.wcat_f  = wcat[s*2+0]; p.bias_f = biasv[s*2+0];
    p.wcat_b  = wcat[s*2+1]; p.bias_b = biasv[s*2+1];
    p.fc_t    = fct[s];
    encoder_kernel<<<1006, 256, 0, stream>>>(s ? gyro : acc, p,
        (const float*)d_in[HEAD + 0], s == 0 ? 1 : 0, seq);
  }

  // MHA head (only CLS row of output is needed)
  kv_gemm_kernel<<<dim3(1008, 3), 256, 0, stream>>>(
      seq, kvw, (const float*)d_in[HEAD + 3], kv);
  attn_cls_kernel<<<dim3(8, 32), 256, 0, stream>>>(
      seq, kv, (const float*)d_in[HEAD + 2], (const float*)d_in[HEAD + 3], attn_cls);
  proj_kernel<<<96, 256, 0, stream>>>(attn_cls, (const float*)d_in[HEAD + 4],
                                      (const float*)d_in[HEAD + 5], tmp);
  proj_kernel<<<96, 256, 0, stream>>>(tmp, (const float*)d_in[HEAD + 6],
                                      (const float*)d_in[HEAD + 7], (float*)d_out);
}
